// Attn_68616397521505
// MI455X (gfx1250) — compile-verified, matrix operand order checked by reading, not run on hardware
//
#include <hip/hip_runtime.h>

#define S_LEN 512
#define BATCH 64
#define HDIM  1024
#define AS_STRIDE 1032   // 1024 + 8 bf16 pad: keeps rows 16B-aligned, staggers banks

typedef __attribute__((ext_vector_type(16))) __bf16 v16bf;
typedef __attribute__((ext_vector_type(8)))  float  v8f;

union Frag { v16bf v; uint4 q[2]; };

__device__ __forceinline__ unsigned short f2bf(float f) {
  unsigned u = __float_as_uint(f);
  u += 0x7FFFu + ((u >> 16) & 1u);          // round-to-nearest-even
  return (unsigned short)(u >> 16);
}

// t[b][h] = ba[h] + sum_k Wa[h][k] * hidden[b][k]   (Wa_h part, s-independent)
__global__ __launch_bounds__(256) void precompute_t(
    const float* __restrict__ hidden, const float* __restrict__ Wa,
    const float* __restrict__ ba, float* __restrict__ t) {
  const int h = blockIdx.x * 256 + threadIdx.x;   // grid.x = 4
  const int b = blockIdx.y;                       // grid.y = 64
  const float4* wa4 = (const float4*)(Wa + (size_t)h * 2 * HDIM);
  const float4* hd4 = (const float4*)(hidden + (size_t)b * HDIM);
  float acc = 0.f;
  #pragma unroll 4
  for (int k = 0; k < HDIM / 4; ++k) {
    float4 w = wa4[k], x = hd4[k];
    acc += w.x * x.x + w.y * x.y + w.z * x.z + w.w * x.w;
  }
  t[(size_t)b * HDIM + h] = acc + ba[h];
}

// Wa_e -> bf16, swizzled [k>>3][n][k&7] so B fragments are contiguous b128 loads
__global__ __launch_bounds__(256) void convert_We(
    const float* __restrict__ Wa, unsigned short* __restrict__ Wb) {
  const int idx = blockIdx.x * 256 + threadIdx.x;   // grid = 4096
  const int n = idx >> 10, k = idx & 1023;
  const float w = Wa[(size_t)n * (2 * HDIM) + HDIM + k];
  Wb[(size_t)((k >> 3) * HDIM + n) * 8 + (k & 7)] = f2bf(w);
}

// Fused: e = relu(Wa_e @ enc[s,:,:]^T + t), logits[b][s] = scale * (e . Ws)
__global__ __launch_bounds__(256) void attn_gemm(
    const float* __restrict__ enc, const unsigned short* __restrict__ Wb,
    const float* __restrict__ t, const float* __restrict__ Ws,
    float* __restrict__ logits) {
  extern __shared__ char smem[];
  unsigned short* As = (unsigned short*)smem;                       // [64][AS_STRIDE] bf16
  float* partial = (float*)(smem + BATCH * AS_STRIDE * sizeof(unsigned short));

  const int tid = threadIdx.x;
  const int s   = blockIdx.x;                                       // one s per workgroup

  // ---- Stage A tile (64 rows x 1024 K) f32 -> bf16 in LDS, coalesced float4 reads
  const float4* enc4 = (const float4*)(enc + (size_t)s * BATCH * HDIM);
  for (int i = tid; i < BATCH * (HDIM / 4); i += 256) {
    const int row = i >> 8;          // HDIM/4 = 256 float4 per row
    const int c4  = i & 255;
    float4 v = enc4[row * 256 + c4];
    ushort4 o;
    o.x = f2bf(v.x); o.y = f2bf(v.y); o.z = f2bf(v.z); o.w = f2bf(v.w);
    *(ushort4*)(As + row * AS_STRIDE + c4 * 4) = o;
  }
  if (tid < BATCH) partial[tid] = 0.f;
  __syncthreads();

  const int lane = tid & 31;
  const int wv   = tid >> 5;          // 8 waves
  const int half = (lane < 16) ? 0 : 1;
  const int l16  = lane & 15;

  for (int nt = wv; nt < HDIM / 16; nt += 8) {
    const int n_g   = nt * 16 + l16;
    const float wsv = Ws[n_g];
    v8f acc[4];
    #pragma unroll
    for (int mt = 0; mt < 4; ++mt) acc[mt] = (v8f){0.f,0.f,0.f,0.f,0.f,0.f,0.f,0.f};

    for (int k = 0; k < HDIM; k += 32) {
      const int kb = k + half * 8;    // lanes 0-15: K 0..7/16..23; 16-31: K 8..15/24..31
      Frag bf_;
      bf_.q[0] = *(const uint4*)(Wb + (size_t)(((kb     ) >> 3) * HDIM + n_g) * 8);
      bf_.q[1] = *(const uint4*)(Wb + (size_t)(((kb + 16) >> 3) * HDIM + n_g) * 8);
      #pragma unroll
      for (int mt = 0; mt < 4; ++mt) {
        Frag af;
        const unsigned short* ap = As + (mt * 16 + l16) * AS_STRIDE + kb;
        af.q[0] = *(const uint4*)ap;           // ds_load_b128: K kb..kb+7
        af.q[1] = *(const uint4*)(ap + 16);    // ds_load_b128: K kb+16..kb+23
        acc[mt] = __builtin_amdgcn_wmma_f32_16x16x32_bf16(
            false, af.v, false, bf_.v, (short)0, acc[mt], false, false);
      }
    }

    // ---- Epilogue: bias+relu, weight by Ws, reduce the 16 n-lanes per row
    #pragma unroll
    for (int mt = 0; mt < 4; ++mt) {
      #pragma unroll
      for (int j = 0; j < 8; ++j) {
        const int b = mt * 16 + half * 8 + j;      // C layout: VGPR j -> M=j / M=8+j
        float v = acc[mt][j] + t[(size_t)b * HDIM + n_g];
        v = fmaxf(v, 0.f) * wsv;
        v += __shfl_xor(v, 8);                     // butterfly within each 16-lane half
        v += __shfl_xor(v, 4);
        v += __shfl_xor(v, 2);
        v += __shfl_xor(v, 1);
        if (l16 == 0) atomicAdd(&partial[b], v);   // ds_add_f32
      }
    }
  }
  __syncthreads();

  if (tid < BATCH) {
    const float scale = 0.19494764453248463f;      // log(512)/sqrt(1024)
    logits[(size_t)tid * S_LEN + s] = partial[tid] * scale;
  }
}

// softmax over S per batch row, with pe add + mask
__global__ __launch_bounds__(256) void softmax_k(
    const float* __restrict__ logits, const float* __restrict__ pe,
    const unsigned char* __restrict__ mask, float* __restrict__ out) {
  const int b  = blockIdx.x;
  const int tx = threadIdx.x;
  __shared__ float red[8];
  const size_t base = (size_t)b * S_LEN;

  float x0 = logits[base + tx]       + pe[base + tx];
  float x1 = logits[base + tx + 256] + pe[base + tx + 256];
  if (mask[base + tx])       x0 = -1e12f;
  if (mask[base + tx + 256]) x1 = -1e12f;

  float m = fmaxf(x0, x1);
  #pragma unroll
  for (int off = 16; off >= 1; off >>= 1) m = fmaxf(m, __shfl_xor(m, off));
  const int lane = tx & 31, w = tx >> 5;
  if (lane == 0) red[w] = m;
  __syncthreads();
  float mm = red[0];
  #pragma unroll
  for (int i = 1; i < 8; ++i) mm = fmaxf(mm, red[i]);

  const float e0 = expf(x0 - mm);
  const float e1 = expf(x1 - mm);
  float ssum = e0 + e1;
  #pragma unroll
  for (int off = 16; off >= 1; off >>= 1) ssum += __shfl_xor(ssum, off);
  __syncthreads();
  if (lane == 0) red[w] = ssum;
  __syncthreads();
  float tot = red[0];
  #pragma unroll
  for (int i = 1; i < 8; ++i) tot += red[i];
  const float inv = 1.0f / tot;

  out[base + tx]       = e0 * inv;     // output (B,1,S) flat = b*512 + s
  out[base + tx + 256] = e1 * inv;
}

extern "C" void kernel_launch(void* const* d_in, const int* in_sizes, int n_in,
                              void* d_out, int out_size, void* d_ws, size_t ws_size,
                              hipStream_t stream) {
  const float* hidden        = (const float*)d_in[0];          // (1,B,H)
  const float* enc           = (const float*)d_in[1];          // (S,B,H)
  const float* pe            = (const float*)d_in[2];          // (B,S)
  const unsigned char* mask  = (const unsigned char*)d_in[3];  // (B,S) bool
  const float* Wa            = (const float*)d_in[4];          // (H,2H)
  const float* ba            = (const float*)d_in[5];          // (H,)
  const float* Ws            = (const float*)d_in[6];          // (1,H)
  float* out = (float*)d_out;

  char* ws = (char*)d_ws;
  float*          t      = (float*)ws;                                      // 256 KB
  unsigned short* Wb     = (unsigned short*)(ws + BATCH * HDIM * 4);        // 2 MB
  float*          logits = (float*)(ws + BATCH * HDIM * 4 + HDIM * HDIM * 2); // 128 KB

  precompute_t<<<dim3(HDIM / 256, BATCH), 256, 0, stream>>>(hidden, Wa, ba, t);
  convert_We<<<(HDIM * HDIM) / 256, 256, 0, stream>>>(Wa, Wb);

  const size_t smem = BATCH * AS_STRIDE * sizeof(unsigned short) + BATCH * sizeof(float);
  attn_gemm<<<S_LEN, 256, smem, stream>>>(enc, Wb, t, Ws, logits);

  softmax_k<<<BATCH, 256, 0, stream>>>(logits, pe, mask, out);
}
